// ProbCrossAttention_88081189306691
// MI455X (gfx1250) — matrix operand order repaired and checked by reading, not compiled, not run on hardware
//
#include <hip/hip_runtime.h>

// ---------------------------------------------------------------------------
// Types / helpers
// ---------------------------------------------------------------------------
typedef __attribute__((ext_vector_type(16))) __bf16 bf16x16;
typedef __attribute__((ext_vector_type(8)))  __bf16 bf16x8;
typedef __attribute__((ext_vector_type(8)))  float  v8f;

union FragAB { bf16x16 v; bf16x8 h[2]; };

__device__ __forceinline__ v8f wmma_bf16(bf16x16 a, bf16x16 b, v8f c) {
  // D = A(16x32 bf16) * B(32x16 bf16) + C(16x16 f32)
  return __builtin_amdgcn_wmma_f32_16x16x32_bf16(false, a, false, b,
                                                 (short)0, c, false, false);
}

// Async global->LDS 16B-per-lane copy (CDNA5, tracked by ASYNCcnt).
// lds_off: low 32 bits of the generic shared-memory address (= LDS offset).
__device__ __forceinline__ void async_ld16(unsigned lds_off, const void* gaddr) {
  asm volatile("global_load_async_to_lds_b128 %0, %1, off"
               :: "v"(lds_off), "v"(gaddr) : "memory");
}
__device__ __forceinline__ unsigned lds_addr(const void* p) {
  return (unsigned)(size_t)p;   // aperture scheme: addr[31:0] = LDS offset
}

constexpr int   kBatch = 8;
constexpr int   kTq    = 1024;
constexpr int   kTk    = 2048;
constexpr int   kC     = 768;
constexpr int   kMq    = kBatch * kTq;   // 8192
constexpr int   kMk    = kBatch * kTk;   // 16384
constexpr float kBeta  = 2.35f;
constexpr float kVarEps = 1e-6f;
constexpr float kLnEps  = 1e-5f;

__device__ __forceinline__ float softplusf(float x) {
  return (x > 20.f) ? x : log1pf(__expf(x));
}

enum { EPI_F32 = 0, EPI_QSQ = 1, EPI_SOFTPLUS = 2, EPI_SQRTSP = 3,
       EPI_BF16 = 4, EPI_GATE = 5 };

// ---------------------------------------------------------------------------
// Generic WMMA GEMM:  Out[m,n] = A[m,:] dot B + bias[n], various epilogues.
//   A: row-major MxK (f32 or bf16), ld = K
//   B: if B_KN: bf16 row-major KxN (ld=N, batched)   [used for P @ V]
//      else   : f32  row-major NxK (ld=K, weights, shared across batch)
// Block: 256 threads = 8 waves (4x2), tile 64x64, K-step 32.
// When A_BF16 && B_KN, staging uses double-buffered GLOBAL_LOAD_ASYNC_TO_LDS.
// ---------------------------------------------------------------------------
template<int EPI, bool A_BF16, bool B_KN>
__global__ __launch_bounds__(256)
void gemm_wmma_kernel(const void* __restrict__ Ap, const void* __restrict__ Bp,
                      const float* __restrict__ bias,
                      void* __restrict__ out0, void* __restrict__ out1,
                      const float* __restrict__ resid,
                      const float* __restrict__ gatep,
                      int M, int N, int K,
                      long batchA, long batchB, long batchO)
{
  constexpr int BM = 64, BN = 64, BK = 32;
  __shared__ alignas(32) __bf16 As[2][BM][BK + 8];   // 80 B rows, 16-B cols
  __shared__ alignas(32) __bf16 Bs[2][BK][BN + 16];  // 160 B rows, 32-B cols

  const int tid  = threadIdx.x;
  const int wave = tid >> 5;
  const int lane = tid & 31;
  const int wr   = wave >> 1;          // 0..3  (M sub-tile)
  const int wc   = wave & 1;           // 0..1  (N sub-tile)
  const int ml   = lane & 15;
  const int hh   = lane >> 4;
  const int z    = blockIdx.z;
  const int m0   = blockIdx.y * BM;
  const int n0   = blockIdx.x * BN;

  const float*  Af = (const float*)Ap  + (size_t)z * batchA;
  const __bf16* Ab = (const __bf16*)Ap + (size_t)z * batchA;
  const float*  Bf = (const float*)Bp;                       // weights (N,K)
  const __bf16* Bb = (const __bf16*)Bp + (size_t)z * batchB; // V (K,N)

  v8f acc0 = {}; v8f acc1 = {};

  if constexpr (A_BF16 && B_KN) {
    // ---------- async double-buffered pipeline (bf16 A and B) ----------
    const int c   = tid * 8;
    const int ar  = c >> 5, ak = c & 31;   // A: row 0..63, k 0..31 (16 B chunk)
    const int bk_ = c >> 6, bn = c & 63;   // B: k 0..31, n 0..63  (16 B chunk)
    const int ntiles = K / BK;

    auto issue = [&](int i, int buf) {
      async_ld16(lds_addr(&As[buf][ar][ak]),
                 Ab + (size_t)(m0 + ar) * K + i * BK + ak);
      async_ld16(lds_addr(&Bs[buf][bk_][bn]),
                 Bb + (size_t)(i * BK + bk_) * N + n0 + bn);
    };
    issue(0, 0);
    for (int i = 0; i < ntiles; ++i) {
      const int cur = i & 1;
      if (i + 1 < ntiles) {
        issue(i + 1, cur ^ 1);
        asm volatile("s_wait_asynccnt 0x2" ::: "memory");  // tile i landed
      } else {
        asm volatile("s_wait_asynccnt 0x0" ::: "memory");
      }
      __syncthreads();
      FragAB a;
      a.h[0] = *(const bf16x8*)&As[cur][16 * wr + ml][8 * hh];
      a.h[1] = *(const bf16x8*)&As[cur][16 * wr + ml][16 + 8 * hh];
      bf16x16 b0, b1;
      #pragma unroll
      for (int s = 0; s < 16; ++s) {
        const int kb = (s < 8) ? 8 * hh + s : 16 + 8 * hh + (s - 8);
        b0[s] = Bs[cur][kb][32 * wc + 0 + ml];
        b1[s] = Bs[cur][kb][32 * wc + 16 + ml];
      }
      acc0 = wmma_bf16(a.v, b0, acc0);
      acc1 = wmma_bf16(a.v, b1, acc1);
      __syncthreads();
    }
  } else {
    // ---------- synchronous staging with f32->bf16 conversion ----------
    for (int k0 = 0; k0 < K; k0 += BK) {
      {
        const int c  = tid * 8;
        const int r  = c >> 5;
        const int kk = c & 31;
        if constexpr (A_BF16) {
          bf16x8 v = *(const bf16x8*)(Ab + (size_t)(m0 + r) * K + k0 + kk);
          *(bf16x8*)&As[0][r][kk] = v;
          if (k0 + BK < K)
            __builtin_prefetch(Ab + (size_t)(m0 + r) * K + k0 + BK + kk, 0, 0);
        } else {
          const float* src = Af + (size_t)(m0 + r) * K + k0 + kk;
          bf16x8 v;
          #pragma unroll
          for (int i = 0; i < 8; ++i) v[i] = (__bf16)src[i];
          *(bf16x8*)&As[0][r][kk] = v;
          if (k0 + BK < K) __builtin_prefetch(src + BK, 0, 0);
        }
      }
      {
        const int c  = tid * 8;
        const int n  = c >> 5;
        const int kk = c & 31;
        const float* src = Bf + (size_t)(n0 + n) * K + k0 + kk;
        #pragma unroll
        for (int i = 0; i < 8; ++i) Bs[0][kk + i][n] = (__bf16)src[i];
        if (k0 + BK < K) __builtin_prefetch(src + BK, 0, 0);
      }
      __syncthreads();
      FragAB a;
      a.h[0] = *(const bf16x8*)&As[0][16 * wr + ml][8 * hh];
      a.h[1] = *(const bf16x8*)&As[0][16 * wr + ml][16 + 8 * hh];
      bf16x16 b0, b1;
      #pragma unroll
      for (int s = 0; s < 16; ++s) {
        const int kb = (s < 8) ? 8 * hh + s : 16 + 8 * hh + (s - 8);
        b0[s] = Bs[0][kb][32 * wc + 0 + ml];
        b1[s] = Bs[0][kb][32 * wc + 16 + ml];
      }
      acc0 = wmma_bf16(a.v, b0, acc0);
      acc1 = wmma_bf16(a.v, b1, acc1);
      __syncthreads();
    }
  }

  // ---- epilogue ----
  float g = 0.f, omg = 0.f;
  if constexpr (EPI == EPI_GATE) {
    const float gs = 1.f / (1.f + __expf(-gatep[0]));
    g = gs; omg = 1.f - gs;
  }
  #pragma unroll
  for (int j = 0; j < 2; ++j) {
    const v8f acc = j ? acc1 : acc0;
    const int n_g = n0 + 32 * wc + 16 * j + ml;
    const float bv = bias ? bias[n_g] : 0.f;
    #pragma unroll
    for (int r = 0; r < 8; ++r) {
      const int m_g = m0 + 16 * wr + r + 8 * hh;
      const float val = acc[r] + bv;
      const size_t idx = (size_t)z * batchO + (size_t)m_g * N + n_g;
      if constexpr (EPI == EPI_F32) {
        ((float*)out0)[idx] = val;
      } else if constexpr (EPI == EPI_QSQ) {
        ((__bf16*)out0)[idx] = (__bf16)val;
        ((__bf16*)out1)[idx] = (__bf16)(val * val);
      } else if constexpr (EPI == EPI_SOFTPLUS) {
        ((__bf16*)out0)[idx] = (__bf16)(softplusf(val) + kVarEps);
      } else if constexpr (EPI == EPI_SQRTSP) {
        ((__bf16*)out0)[idx] = (__bf16)sqrtf(softplusf(val) + kVarEps);
      } else if constexpr (EPI == EPI_BF16) {
        ((__bf16*)out0)[idx] = (__bf16)val;
      } else if constexpr (EPI == EPI_GATE) {
        ((float*)out0)[idx] = g * val + omg * resid[idx];
      }
    }
  }
}

// ---------------------------------------------------------------------------
// Row LayerNorm (rows of 768), optional reparameterized sampling epilogue.
//   SAMPLE=0: out = bf16(LN(x)*g + b)                        (K_mu)
//   SAMPLE=1: out = bf16(LN(x)*g + b + sdev*noise)           (V_sample)
// ---------------------------------------------------------------------------
template<bool SAMPLE>
__global__ __launch_bounds__(256)
void ln_kernel(const float* __restrict__ X,
               const float* __restrict__ gam, const float* __restrict__ bet,
               const __bf16* __restrict__ sdev, const float* __restrict__ noise,
               __bf16* __restrict__ out)
{
  __shared__ float red[256];
  const int row = blockIdx.x;
  const int tid = threadIdx.x;
  const float* x = X + (size_t)row * kC;

  const float v0 = x[tid], v1 = x[tid + 256], v2 = x[tid + 512];
  red[tid] = v0 + v1 + v2;
  __syncthreads();
  for (int o = 128; o > 0; o >>= 1) { if (tid < o) red[tid] += red[tid + o]; __syncthreads(); }
  const float mean = red[0] * (1.f / kC);
  __syncthreads();

  const float d0 = v0 - mean, d1 = v1 - mean, d2 = v2 - mean;
  red[tid] = d0 * d0 + d1 * d1 + d2 * d2;
  __syncthreads();
  for (int o = 128; o > 0; o >>= 1) { if (tid < o) red[tid] += red[tid + o]; __syncthreads(); }
  const float rstd = rsqrtf(red[0] * (1.f / kC) + kLnEps);

  #pragma unroll
  for (int p = 0; p < 3; ++p) {
    const int col = tid + 256 * p;
    const float v = (p == 0) ? v0 : (p == 1) ? v1 : v2;
    float y = (v - mean) * rstd * gam[col] + bet[col];
    if constexpr (SAMPLE) {
      const size_t idx = (size_t)row * kC + col;
      y += (float)sdev[idx] * noise[idx];
    }
    out[(size_t)row * kC + col] = (__bf16)y;
  }
}

// ---------------------------------------------------------------------------
// Fused probabilistic attention scores + 2-pass online softmax -> P (bf16).
// Computes S^T tiles: A = Kmu/Kvar rows (global), B = Q^T / (Q^2)^T from LDS.
// D layout: lane%16 = query, reg r -> key r+8*(lane/16).  Softmax stats are
// therefore per-lane; merged across halves via shfl and across waves via LDS.
// ---------------------------------------------------------------------------
__device__ __forceinline__ void score_tile(
    const __bf16* __restrict__ kmu_row, const __bf16* __restrict__ kvar_row,
    const __bf16 (*Qts)[16], const __bf16 (*Q2ts)[16],
    int lane, int hh, v8f& sm, v8f& sv)
{
  for (int c0 = 0; c0 < kC; c0 += 32) {
    FragAB am, av;
    am.h[0] = *(const bf16x8*)(kmu_row  + c0 + 8 * hh);
    am.h[1] = *(const bf16x8*)(kmu_row  + c0 + 16 + 8 * hh);
    av.h[0] = *(const bf16x8*)(kvar_row + c0 + 8 * hh);
    av.h[1] = *(const bf16x8*)(kvar_row + c0 + 16 + 8 * hh);
    bf16x16 bm, bv;
    #pragma unroll
    for (int s = 0; s < 16; ++s) {
      const int kb = (s < 8) ? 8 * hh + s : 16 + 8 * hh + (s - 8);
      bm[s] = Qts[c0 + kb][lane & 15];
      bv[s] = Q2ts[c0 + kb][lane & 15];
    }
    sm = wmma_bf16(am.v, bm, sm);
    sv = wmma_bf16(av.v, bv, sv);
  }
}

__global__ __launch_bounds__(256)
void attn_scores_kernel(const __bf16* __restrict__ Qb,  const __bf16* __restrict__ Q2b,
                        const __bf16* __restrict__ Kmu, const __bf16* __restrict__ Kvar,
                        __bf16* __restrict__ P)
{
  constexpr int NW = 8;                       // waves per block
  __shared__ alignas(32) __bf16 Qts[kC][16];  // Q^T   tile (24 KB)
  __shared__ alignas(32) __bf16 Q2ts[kC][16]; // Q^2^T tile (24 KB)
  __shared__ float smax[NW][16], ssum[NW][16];
  __shared__ float fm[16], fl[16];

  const int tid = threadIdx.x, wave = tid >> 5, lane = tid & 31;
  const int ml = lane & 15, hh = lane >> 4;
  const int b  = blockIdx.x >> 6;             // Tq/16 = 64 q-tiles per batch
  const int q0 = (blockIdx.x & 63) * 16;

  // stage Q, Q^2 transposed into LDS
  for (int c8 = tid; c8 < 16 * (kC / 8); c8 += 256) {
    const int m  = c8 / (kC / 8);
    const int cc = (c8 % (kC / 8)) * 8;
    const size_t base = ((size_t)(b * kTq + q0 + m)) * kC + cc;
    bf16x8 v = *(const bf16x8*)(Qb + base);
    bf16x8 w = *(const bf16x8*)(Q2b + base);
    #pragma unroll
    for (int i = 0; i < 8; ++i) { Qts[cc + i][m] = v[i]; Q2ts[cc + i][m] = w[i]; }
  }
  __syncthreads();

  const float inv_sqrtC = rsqrtf((float)kC);
  const float invC      = 1.f / (float)kC;

  // ---- pass A: online (m, l) per lane ----
  float m_run = -3.4e38f, l_run = 0.f;
  for (int kt = wave; kt < kTk / 16; kt += NW) {
    const __bf16* kmu_row  = Kmu  + ((size_t)(b * kTk + kt * 16 + ml)) * kC;
    const __bf16* kvar_row = Kvar + ((size_t)(b * kTk + kt * 16 + ml)) * kC;
    v8f sm = {}, sv = {};
    score_tile(kmu_row, kvar_row, Qts, Q2ts, lane, hh, sm, sv);
    #pragma unroll
    for (int r = 0; r < 8; ++r) {
      const float s = sm[r] * inv_sqrtC - kBeta * sqrtf(fmaxf(sv[r], 0.f) * invC);
      const float mn = fmaxf(m_run, s);
      l_run = l_run * __expf(m_run - mn) + __expf(s - mn);
      m_run = mn;
    }
  }
  // merge the two key-halves (lane <-> lane+16, same query)
  {
    const float mo = __shfl_xor(m_run, 16);
    const float lo = __shfl_xor(l_run, 16);
    const float mn = fmaxf(m_run, mo);
    l_run = l_run * __expf(m_run - mn) + lo * __expf(mo - mn);
    m_run = mn;
  }
  if (lane < 16) { smax[wave][lane] = m_run; ssum[wave][lane] = l_run; }
  __syncthreads();
  if (tid < 16) {
    float M = -3.4e38f;
    for (int w = 0; w < NW; ++w) M = fmaxf(M, smax[w][tid]);
    float L = 0.f;
    for (int w = 0; w < NW; ++w) L += ssum[w][tid] * __expf(smax[w][tid] - M);
    fm[tid] = M; fl[tid] = 1.f / L;
  }
  __syncthreads();
  const float Mf = fm[ml];
  const float Li = fl[ml];

  // ---- pass B: recompute tiles, emit normalized P ----
  for (int kt = wave; kt < kTk / 16; kt += NW) {
    const __bf16* kmu_row  = Kmu  + ((size_t)(b * kTk + kt * 16 + ml)) * kC;
    const __bf16* kvar_row = Kvar + ((size_t)(b * kTk + kt * 16 + ml)) * kC;
    v8f sm = {}, sv = {};
    score_tile(kmu_row, kvar_row, Qts, Q2ts, lane, hh, sm, sv);
    #pragma unroll
    for (int r = 0; r < 8; ++r) {
      const float s = sm[r] * inv_sqrtC - kBeta * sqrtf(fmaxf(sv[r], 0.f) * invC);
      const float p = __expf(s - Mf) * Li;
      const int key = kt * 16 + r + 8 * hh;
      P[((size_t)(b * kTq + q0 + ml)) * kTk + key] = (__bf16)p;
    }
  }
}

// ---------------------------------------------------------------------------
// Host-side launcher
// ---------------------------------------------------------------------------
extern "C" void kernel_launch(void* const* d_in, const int* in_sizes, int n_in,
                              void* d_out, int out_size, void* d_ws, size_t ws_size,
                              hipStream_t stream) {
  (void)in_sizes; (void)n_in; (void)out_size; (void)ws_size;
  const float* query   = (const float*)d_in[0];
  const float* context = (const float*)d_in[1];
  const float* noise   = (const float*)d_in[2];
  const float* Wq      = (const float*)d_in[3];
  const float* bq      = (const float*)d_in[4];
  const float* Wk      = (const float*)d_in[5];
  const float* bk      = (const float*)d_in[6];
  const float* Wv      = (const float*)d_in[7];
  const float* bv      = (const float*)d_in[8];
  const float* Wo      = (const float*)d_in[9];
  const float* bo      = (const float*)d_in[10];
  const float* ln_k_g  = (const float*)d_in[11];
  const float* ln_k_b  = (const float*)d_in[12];
  const float* ln_v_g  = (const float*)d_in[13];
  const float* ln_v_b  = (const float*)d_in[14];
  const float* gate    = (const float*)d_in[15];
  float* out = (float*)d_out;

  // workspace layout (lifetime-aliased)
  char* ws = (char*)d_ws;
  size_t o = 0;
  __bf16* Qbf   = (__bf16*)(ws + o); o += (size_t)kMq * kC * 2;  // 12.6 MB
  __bf16* Q2bf  = (__bf16*)(ws + o); o += (size_t)kMq * kC * 2;  // 12.6 MB
  __bf16* KmuB  = (__bf16*)(ws + o); o += (size_t)kMk * kC * 2;  // 25.2 MB
  __bf16* KvarB = (__bf16*)(ws + o); o += (size_t)kMk * kC * 2;  // 25.2 MB
  __bf16* VsB   = (__bf16*)(ws + o); o += (size_t)kMk * kC * 2;  // 25.2 MB
  // region R2: sqrtVvar (bf16, MkxC) then attn-out (bf16, MqxC)
  __bf16* sVv   = (__bf16*)(ws + o);
  __bf16* AoB   = (__bf16*)(ws + o); o += (size_t)kMk * kC * 2;  // 25.2 MB
  // region R1: raw f32 (MkxC) for LN inputs, later reused for P (bf16)
  float*  raw   = (float*)(ws + o);
  __bf16* P     = (__bf16*)(ws + o);                             // 50.3 MB

  const dim3 blk(256);
  const dim3 gQ (kC / 64, kMq / 64, 1);
  const dim3 gK (kC / 64, kMk / 64, 1);
  const dim3 gPV(kC / 64, kTq / 64, kBatch);

  // 1. Q projection -> Qbf, Q2bf
  gemm_wmma_kernel<EPI_QSQ, false, false><<<gQ, blk, 0, stream>>>(
      query, Wq, bq, Qbf, Q2bf, nullptr, nullptr, kMq, kC, kC, 0, 0, 0);
  // 2. K_mu raw projection
  gemm_wmma_kernel<EPI_F32, false, false><<<gK, blk, 0, stream>>>(
      context, Wk, bk, raw, nullptr, nullptr, nullptr, kMk, kC, kC, 0, 0, 0);
  // 3. LayerNorm K_mu -> bf16
  ln_kernel<false><<<dim3(kMk), blk, 0, stream>>>(raw, ln_k_g, ln_k_b,
                                                  nullptr, nullptr, KmuB);
  // 4. K_var = softplus + eps (fused epilogue)
  gemm_wmma_kernel<EPI_SOFTPLUS, false, false><<<gK, blk, 0, stream>>>(
      context, Wk + (size_t)kC * kC, bk + kC, KvarB, nullptr, nullptr, nullptr,
      kMk, kC, kC, 0, 0, 0);
  // 5. V_mu raw projection (reuses raw)
  gemm_wmma_kernel<EPI_F32, false, false><<<gK, blk, 0, stream>>>(
      context, Wv, bv, raw, nullptr, nullptr, nullptr, kMk, kC, kC, 0, 0, 0);
  // 6. sqrt(V_var) (fused epilogue)
  gemm_wmma_kernel<EPI_SQRTSP, false, false><<<gK, blk, 0, stream>>>(
      context, Wv + (size_t)kC * kC, bv + kC, sVv, nullptr, nullptr, nullptr,
      kMk, kC, kC, 0, 0, 0);
  // 7. V_sample = LN(V_mu) + sqrtVvar * noise
  ln_kernel<true><<<dim3(kMk), blk, 0, stream>>>(raw, ln_v_g, ln_v_b,
                                                 sVv, noise, VsB);
  // 8. fused scores + 2-pass softmax -> P (overwrites raw region)
  attn_scores_kernel<<<dim3(kBatch * (kTq / 16)), blk, 0, stream>>>(
      Qbf, Q2bf, KmuB, KvarB, P);
  // 9. attn @ V_sample (batched, async double-buffered LDS) -> attn-out bf16
  gemm_wmma_kernel<EPI_BF16, true, true><<<gPV, blk, 0, stream>>>(
      P, VsB, nullptr, AoB, nullptr, nullptr, nullptr,
      kTq, kC, kTk, (long)kTq * kTk, (long)kTk * kC, (long)kTq * kC);
  // 10. output projection + sigmoid-gated residual -> d_out (f32)
  gemm_wmma_kernel<EPI_GATE, true, false><<<gQ, blk, 0, stream>>>(
      AoB, Wo, bo, out, nullptr, query, gate, kMq, kC, kC, 0, 0, 0);
}
